// DiffAttention_83906481095013
// MI455X (gfx1250) — matrix operand order repaired and checked by reading, not compiled, not run on hardware
//
#include <hip/hip_runtime.h>
#include <cstdint>

// ---------------------------------------------------------------------------
// CDNA5 / gfx1250 differential attention
//   - bf16 WMMA (fp32 accumulate) for all 4 GEMMs + both attention einsums
//   - GLOBAL_LOAD_ASYNC_TO_LDS_B128 double-buffered tile staging (ASYNCcnt)
// ---------------------------------------------------------------------------

typedef __attribute__((ext_vector_type(16))) __bf16 v16bf;
typedef __attribute__((ext_vector_type(8)))  float  v8f;

struct alignas(16) U128 { uint32_t a, b, c, d; };
struct alignas(32) U256 { U128 lo, hi; };

__device__ __forceinline__ uint16_t bfbits(float f) {
  uint32_t u = __builtin_bit_cast(uint32_t, f);
  u += 0x7FFFu + ((u >> 16) & 1u);          // round-to-nearest-even
  return (uint16_t)(u >> 16);
}
__device__ __forceinline__ __bf16 f2bf(float f) {
  uint16_t h = bfbits(f);
  return __builtin_bit_cast(__bf16, h);
}
__device__ __forceinline__ uint32_t pack2(float a, float b) {
  return (uint32_t)bfbits(a) | ((uint32_t)bfbits(b) << 16);
}
// 16 contiguous bf16 (per-lane run in contiguous k order, used for A and B)
__device__ __forceinline__ v16bf ld_frag16(const __bf16* p) {
  U256 u; u.lo = *(const U128*)(p); u.hi = *(const U128*)(p + 8);
  return __builtin_bit_cast(v16bf, u);
}
// A-operand per-lane run: 8 bf16 at p, 8 bf16 at p+16
__device__ __forceinline__ v16bf ld_frag_a(const __bf16* p) {
  U256 u; u.lo = *(const U128*)(p); u.hi = *(const U128*)(p + 16);
  return __builtin_bit_cast(v16bf, u);
}
__device__ __forceinline__ v8f zero8() {
  v8f z; for (int i = 0; i < 8; ++i) z[i] = 0.f; return z;
}
// CDNA5 async copy: 16B global -> LDS, tracked by ASYNCcnt (no VGPR round trip)
__device__ __forceinline__ void async_cp16(uint32_t lds_off, const void* gptr) {
  asm volatile("global_load_async_to_lds_b128 %0, %1, off"
               :: "v"(lds_off), "v"((unsigned long long)(uintptr_t)gptr)
               : "memory");
}

static constexpr int NN = 2048;   // M = N = K = EMBED = SEQ

// ---------------------------------------------------------------------------
// bulk fp32 -> bf16 pack (one-time; removes conversions from GEMM hot loops)
// ---------------------------------------------------------------------------
__global__ void pack_bf16(const float* __restrict__ in, __bf16* __restrict__ outp) {
  const int tid = blockIdx.x * blockDim.x + threadIdx.x;   // < F/4
  const float4 v = *(const float4*)(in + (size_t)tid * 4);
  uint2 u; u.x = pack2(v.x, v.y); u.y = pack2(v.z, v.w);
  *(uint2*)(outp + (size_t)tid * 4) = u;
}

// ---------------------------------------------------------------------------
// GEMM: C[M][N] = A[M][K] * B[N][K]^T, bf16 operands in HBM, fp32 out.
// 128x128 block tile, BK=64, async double-buffered LDS staging.
// 8 waves (2 x 4), 4x2 WMMA tiles per wave, 16 WMMAs per wave per K-tile.
// ---------------------------------------------------------------------------
__global__ __launch_bounds__(256) void gemm_bf16_nt(const __bf16* __restrict__ A,
                                                    const __bf16* __restrict__ B,
                                                    float* __restrict__ C) {
  __shared__ uint32_t As[2][128 * 32];   // 128 rows x 64 k bf16 = 16KB per buf
  __shared__ uint32_t Bs[2][128 * 32];

  const int tid  = threadIdx.x;
  const int lane = tid & 31;
  const int w    = tid >> 5;
  const int wm   = w & 1;        // 0..1 -> 64-row slab
  const int wn   = w >> 1;       // 0..3 -> 32-col slab
  const int lh   = lane & 15;
  const int hb   = lane >> 4;    // lane half
  const int rowBase = blockIdx.x * 128;
  const int colBase = blockIdx.y * 128;

  // issue one 128x64 bf16 tile pair: 8 async b128 copies per thread
  auto issue_tile = [&](int k0, int buf) {
    for (int i = 0; i < 4; ++i) {
      const int e = tid + 256 * i;   // 16B-chunk id in [0,1024)
      const int r = e >> 2;          // 4 x 16B chunks per 128B row
      const int c = e & 3;
      async_cp16((uint32_t)(uintptr_t)&As[buf][r * 32 + c * 4],
                 A + (size_t)(rowBase + r) * NN + k0 + c * 8);
      async_cp16((uint32_t)(uintptr_t)&Bs[buf][r * 32 + c * 4],
                 B + (size_t)(colBase + r) * NN + k0 + c * 8);
    }
  };

  v8f acc[4][2];
  for (int i = 0; i < 4; ++i)
    for (int j = 0; j < 2; ++j) acc[i][j] = zero8();

  issue_tile(0, 0);
  for (int it = 0; it < NN / 64; ++it) {
    const int buf = it & 1;
    if (it + 1 < NN / 64) {
      issue_tile((it + 1) * 64, buf ^ 1);
      asm volatile("s_wait_asynccnt 0x8" ::: "memory");  // previous batch done
    } else {
      asm volatile("s_wait_asynccnt 0x0" ::: "memory");
    }
    __syncthreads();

    const uint32_t* as = As[buf];
    const uint32_t* bs = Bs[buf];
    for (int ks = 0; ks < 2; ++ks) {
      v16bf afr[4], bfr[2];
      const int ab2 = ks * 16 + (hb ? 8 : 0);   // A: k base 0/16 bf16 -> /2 uints
      for (int mt = 0; mt < 4; ++mt) {
        const int r = wm * 64 + mt * 16 + lh;
        U256 u;
        u.lo = *(const U128*)&as[r * 32 + ab2];
        u.hi = *(const U128*)&as[r * 32 + ab2 + 4];
        afr[mt] = __builtin_bit_cast(v16bf, u);
      }
      const int kb2 = ks * 16 + (hb ? 8 : 0);   // B: k base 0/16 bf16 -> /2
      for (int nt = 0; nt < 2; ++nt) {
        const int cidx = wn * 32 + nt * 16 + lh;
        U256 u;
        u.lo = *(const U128*)&bs[cidx * 32 + kb2];
        u.hi = *(const U128*)&bs[cidx * 32 + kb2 + 4];
        bfr[nt] = __builtin_bit_cast(v16bf, u);
      }
      for (int mt = 0; mt < 4; ++mt)
        for (int nt = 0; nt < 2; ++nt)
          acc[mt][nt] = __builtin_amdgcn_wmma_f32_16x16x32_bf16(
              false, afr[mt], false, bfr[nt], (short)0, acc[mt][nt], false, false);
    }
    __syncthreads();   // all reads of buf done before it is overwritten
  }

  for (int mt = 0; mt < 4; ++mt)
    for (int nt = 0; nt < 2; ++nt)
      for (int g = 0; g < 8; ++g) {
        const int row = rowBase + wm * 64 + mt * 16 + g + hb * 8;
        const int col = colBase + wn * 32 + nt * 16 + lh;
        C[(size_t)row * NN + col] = acc[mt][nt][g];
      }
}

// ---------------------------------------------------------------------------
// RoPE on q,k + pack to bf16 [nh=32][t=2048][64]. One thread per (t, nh, pair)
// ---------------------------------------------------------------------------
__global__ void rope_pack_qk(const float* __restrict__ qf, const float* __restrict__ kf,
                             __bf16* __restrict__ qb, __bf16* __restrict__ kb) {
  const int tid = blockIdx.x * blockDim.x + threadIdx.x;   // < 2048*32*32
  const int t  = tid >> 10;
  const int nh = (tid >> 5) & 31;
  const int i  = tid & 31;
  const float inv = __expf(-((float)(2 * i) / 64.f) * 9.2103403719761836f);
  const float ang = (float)t * inv;
  float s, c;
  __sincosf(ang, &s, &c);
  const size_t src = (size_t)t * NN + nh * 64 + 2 * i;
  const size_t dst = ((size_t)nh * NN + t) * 64 + 2 * i;
  const float qr = qf[src], qi = qf[src + 1];
  qb[dst]     = f2bf(qr * c - qi * s);
  qb[dst + 1] = f2bf(qr * s + qi * c);
  const float kr = kf[src], ki = kf[src + 1];
  kb[dst]     = f2bf(kr * c - ki * s);
  kb[dst + 1] = f2bf(kr * s + ki * c);
}

// V fp32 [t][h*128+d] -> bf16 transposed [h][d][t] (WMMA B layout for P*V)
__global__ void pack_v(const float* __restrict__ vf, __bf16* __restrict__ vb) {
  const int tid = blockIdx.x * blockDim.x + threadIdx.x;   // < 2048*2048
  const int t = tid >> 11;
  const int j = tid & 2047;
  const int h = j >> 7, d = j & 127;
  vb[((size_t)h * 128 + d) * NN + t] = f2bf(vf[tid]);
}

__global__ void calc_lambda(const float* q1, const float* q2,
                            const float* k1, const float* k2, float* out) {
  if (threadIdx.x == 0 && blockIdx.x == 0) {
    float s1 = 0.f, s2 = 0.f;
    for (int i = 0; i < 64; ++i) { s1 += q1[i] * k1[i]; s2 += q2[i] * k2[i]; }
    out[0] = __expf(s1) - __expf(s2) + 0.7836057665316245f;  // LAMBDA_INIT
  }
}

// ---------------------------------------------------------------------------
// Fused differential flash attention.
// grid = (T/128, H=16); block = 256 (8 waves). Wave owns 16 query rows for
// BOTH component heads 2h, 2h+1. K/V chunks staged once per block via async
// LDS copies (double buffered); online softmax; shared V fragments.
// ---------------------------------------------------------------------------
__global__ __launch_bounds__(256) void diff_attn(
    const __bf16* __restrict__ qb, const __bf16* __restrict__ kb,
    const __bf16* __restrict__ vb, const float* __restrict__ lamp,
    const float* __restrict__ subw, __bf16* __restrict__ attnb) {
  __shared__ __bf16 Klds[2][2 * 32 * 64];   // [buf][hd][s(32)][c(64)]  8KB/buf
  __shared__ __bf16 Vlds[2][128 * 32];      // [buf][d(128)][s(32)]     8KB/buf
  __shared__ __bf16 Plds[8 * 2 * 16 * 32];  // per-wave per-head P tile 16KB

  const int tid  = threadIdx.x;
  const int lane = tid & 31;
  const int w    = tid >> 5;
  const int lh   = lane & 15;
  const int hb   = lane >> 4;
  const int h    = blockIdx.y;
  const int qr0  = blockIdx.x * 128 + w * 16;
  const float lam = lamp[0];
  const float SC  = 11.3137084989847604f;    // 1/SCALING = sqrt(128)

  // stage one 32-wide K/V chunk: 4 async b128 copies per thread
  auto issue_kv = [&](int s0, int buf) {
    for (int i = 0; i < 2; ++i) {            // K: 512 x 16B chunks
      const int e   = tid + 256 * i;
      const int hd  = e >> 8;
      const int row = (e >> 3) & 31;
      const int c8  = e & 7;
      async_cp16((uint32_t)(uintptr_t)&Klds[buf][(hd * 32 + row) * 64 + c8 * 8],
                 kb + ((size_t)(2 * h + hd) * NN + s0 + row) * 64 + c8 * 8);
    }
    for (int i = 0; i < 2; ++i) {            // V: 512 x 16B chunks
      const int e  = tid + 256 * i;
      const int d  = e >> 2;
      const int c4 = e & 3;
      async_cp16((uint32_t)(uintptr_t)&Vlds[buf][d * 32 + c4 * 8],
                 vb + ((size_t)h * 128 + d) * NN + s0 + c4 * 8);
    }
  };

  // Q fragments (A layout), loaded once: 2 heads x 2 k-steps
  v16bf qfr[2][2];
  for (int hd = 0; hd < 2; ++hd) {
    const int nh = 2 * h + hd;
    for (int ks = 0; ks < 2; ++ks)
      qfr[hd][ks] = ld_frag16(qb + ((size_t)nh * NN + qr0 + lh) * 64 + ks * 32 + (hb ? 16 : 0));
  }

  float m[2][8], l[2][8];
  v8f   nacc[2][8];                           // numerators, 8 d-tiles x 16 cols
  for (int hd = 0; hd < 2; ++hd)
    for (int g = 0; g < 8; ++g) { m[hd][g] = -1e30f; l[hd][g] = 0.f; nacc[hd][g] = zero8(); }

  issue_kv(0, 0);
  for (int it = 0; it < NN / 32; ++it) {
    const int buf = it & 1;
    if (it + 1 < NN / 32) {
      issue_kv((it + 1) * 32, buf ^ 1);
      asm volatile("s_wait_asynccnt 0x4" ::: "memory");
    } else {
      asm volatile("s_wait_asynccnt 0x0" ::: "memory");
    }
    __syncthreads();

    for (int hd = 0; hd < 2; ++hd) {
      v8f S[2];
      for (int st = 0; st < 2; ++st) {
        S[st] = zero8();
        for (int ks = 0; ks < 2; ++ks) {
          const v16bf kfr = ld_frag16(
              &Klds[buf][(hd * 32 + st * 16 + lh) * 64 + ks * 32 + (hb ? 16 : 0)]);
          S[st] = __builtin_amdgcn_wmma_f32_16x16x32_bf16(
              false, qfr[hd][ks], false, kfr, (short)0, S[st], false, false);
        }
        for (int g = 0; g < 8; ++g) S[st][g] *= SC;
      }
      // row max over 32 cols: per-element then 16-lane xor reduce
      float cmax[8];
      for (int g = 0; g < 8; ++g) cmax[g] = fmaxf(S[0][g], S[1][g]);
      for (int msk = 1; msk <= 8; msk <<= 1)
        for (int g = 0; g < 8; ++g)
          cmax[g] = fmaxf(cmax[g], __shfl_xor(cmax[g], msk, 32));
      float sf[8];
      for (int g = 0; g < 8; ++g) {
        const float mn = fmaxf(m[hd][g], cmax[g]);
        sf[g] = __expf(m[hd][g] - mn);
        m[hd][g] = mn;
        l[hd][g] *= sf[g];
      }
      for (int dt = 0; dt < 8; ++dt)
        for (int g = 0; g < 8; ++g) nacc[hd][dt][g] *= sf[g];
      // P = exp(S - m); row-sum; spill P to LDS (C layout -> [row][s] tile)
      float psum[8];
      for (int g = 0; g < 8; ++g) psum[g] = 0.f;
      for (int st = 0; st < 2; ++st)
        for (int g = 0; g < 8; ++g) {
          const float p = __expf(S[st][g] - m[hd][g]);
          psum[g] += p;
          Plds[((w * 2 + hd) * 16 + g + hb * 8) * 32 + st * 16 + lh] = f2bf(p);
        }
      for (int msk = 1; msk <= 8; msk <<= 1)
        for (int g = 0; g < 8; ++g) psum[g] += __shfl_xor(psum[g], msk, 32);
      for (int g = 0; g < 8; ++g) l[hd][g] += psum[g];
    }

    asm volatile("s_wait_dscnt 0x0" ::: "memory");  // wave-local LDS RAW fence

    v16bf pa[2];
    for (int hd = 0; hd < 2; ++hd)
      pa[hd] = ld_frag16(&Plds[((w * 2 + hd) * 16 + lh) * 32 + (hb ? 16 : 0)]);

    // P * V: shared V fragment feeds both component-head numerators
    for (int dt = 0; dt < 8; ++dt) {
      const v16bf vfr = ld_frag16(&Vlds[buf][(dt * 16 + lh) * 32 + (hb ? 16 : 0)]);
      nacc[0][dt] = __builtin_amdgcn_wmma_f32_16x16x32_bf16(
          false, pa[0], false, vfr, (short)0, nacc[0][dt], false, false);
      nacc[1][dt] = __builtin_amdgcn_wmma_f32_16x16x32_bf16(
          false, pa[1], false, vfr, (short)0, nacc[1][dt], false, false);
    }
    __syncthreads();   // all reads of buf done before it is overwritten
  }

  // ---- epilogue: diff, RMS norm over d=128, subln weight, store bf16 [t][h*128+d]
  float inv0[8], inv1[8];
  for (int g = 0; g < 8; ++g) { inv0[g] = 1.f / l[0][g]; inv1[g] = lam / l[1][g]; }
  float ss[8];
  for (int g = 0; g < 8; ++g) ss[g] = 0.f;
  for (int dt = 0; dt < 8; ++dt)
    for (int g = 0; g < 8; ++g) {
      const float v = nacc[0][dt][g] * inv0[g] - nacc[1][dt][g] * inv1[g];
      nacc[0][dt][g] = v;
      ss[g] += v * v;
    }
  for (int msk = 1; msk <= 8; msk <<= 1)
    for (int g = 0; g < 8; ++g) ss[g] += __shfl_xor(ss[g], msk, 32);
  float rr[8];
  for (int g = 0; g < 8; ++g) rr[g] = rsqrtf(ss[g] * (1.f / 128.f) + 1e-5f);
  for (int dt = 0; dt < 8; ++dt) {
    const float wd = subw[dt * 16 + lh];
    for (int g = 0; g < 8; ++g) {
      const int row = qr0 + g + hb * 8;
      attnb[(size_t)row * NN + h * 128 + dt * 16 + lh] = f2bf(nacc[0][dt][g] * rr[g] * wd);
    }
  }
}

// ---------------------------------------------------------------------------
extern "C" void kernel_launch(void* const* d_in, const int* in_sizes, int n_in,
                              void* d_out, int out_size, void* d_ws, size_t ws_size,
                              hipStream_t stream) {
  (void)in_sizes; (void)n_in; (void)out_size; (void)ws_size;
  const float* x    = (const float*)d_in[0];
  const float* wq   = (const float*)d_in[1];
  const float* wk   = (const float*)d_in[2];
  const float* wv   = (const float*)d_in[3];
  const float* wout = (const float*)d_in[4];
  const float* lq1  = (const float*)d_in[5];
  const float* lq2  = (const float*)d_in[6];
  const float* lk1  = (const float*)d_in[7];
  const float* lk2  = (const float*)d_in[8];
  const float* subw = (const float*)d_in[9];
  float* out = (float*)d_out;

  const size_t F = (size_t)NN * NN;
  __bf16* xb    = (__bf16*)d_ws;       //  8 MB
  __bf16* wqb   = xb + F;              //  8 MB
  __bf16* wkb   = wqb + F;             //  8 MB
  __bf16* wvb   = wkb + F;             //  8 MB
  __bf16* woutb = wvb + F;             //  8 MB
  float*  qf    = (float*)(woutb + F); // 16 MB
  float*  kf    = qf + F;              // 16 MB
  float*  vf    = kf + F;              // 16 MB
  __bf16* qbp   = (__bf16*)(vf + F);   //  8 MB
  __bf16* kbp   = qbp + F;             //  8 MB
  __bf16* vbp   = kbp + F;             //  8 MB
  __bf16* attnb = vbp + F;             //  8 MB
  float*  lamp  = (float*)(attnb + F); //  4 B

  const dim3 gg(16, 16), gb(256);
  const dim3 pg((unsigned)(F / 4 / 256)), pb(256);
  hipLaunchKernelGGL(pack_bf16, pg, pb, 0, stream, x, xb);
  hipLaunchKernelGGL(pack_bf16, pg, pb, 0, stream, wq, wqb);
  hipLaunchKernelGGL(pack_bf16, pg, pb, 0, stream, wk, wkb);
  hipLaunchKernelGGL(pack_bf16, pg, pb, 0, stream, wv, wvb);
  hipLaunchKernelGGL(pack_bf16, pg, pb, 0, stream, wout, woutb);
  hipLaunchKernelGGL(gemm_bf16_nt, gg, gb, 0, stream, xb, wqb, qf);
  hipLaunchKernelGGL(gemm_bf16_nt, gg, gb, 0, stream, xb, wkb, kf);
  hipLaunchKernelGGL(gemm_bf16_nt, gg, gb, 0, stream, xb, wvb, vf);
  hipLaunchKernelGGL(rope_pack_qk, dim3((NN * 32 * 32) / 256), dim3(256), 0, stream,
                     qf, kf, qbp, kbp);
  hipLaunchKernelGGL(pack_v, dim3((unsigned)(F / 256)), dim3(256), 0, stream, vf, vbp);
  hipLaunchKernelGGL(calc_lambda, dim3(1), dim3(64), 0, stream, lq1, lq2, lk1, lk2, lamp);
  hipLaunchKernelGGL(diff_attn, dim3(16, 16), dim3(256), 0, stream,
                     qbp, kbp, vbp, lamp, subw, attnb);
  hipLaunchKernelGGL(gemm_bf16_nt, gg, gb, 0, stream, attnb, woutb, out);
}
